// UserContextualFeatures_31653908971571
// MI455X (gfx1250) — matrix operand order repaired and checked by reading, not compiled, not run on hardware
//
#include <hip/hip_runtime.h>
#include <hip/hip_bf16.h>

typedef __attribute__((ext_vector_type(16))) __bf16 v16bf;
typedef __attribute__((ext_vector_type(8)))  float  v8f;
typedef __attribute__((ext_vector_type(4)))  float  f32x4;
typedef __attribute__((ext_vector_type(2)))  float  f32x2;

#define GLB __attribute__((address_space(1)))

#define B_ 128
#define T_ 512
#define V_ 50000
#define D_ 512

// Pack two f32 into two bf16 (round-half-up): two v_add + one v_perm_b32.
// Result dword = { bf16(hi), bf16(lo) } (lo in bits [15:0]).
__device__ __forceinline__ unsigned pkbf(float lo, float hi) {
    unsigned a = __float_as_uint(lo) + 0x8000u;
    unsigned b = __float_as_uint(hi) + 0x8000u;
    return __builtin_amdgcn_perm(b, a, 0x07060302u);
}

__global__ void ucf_zero_kernel(float* __restrict__ p, long n4) {
    GLB f32x4* pg = (GLB f32x4*)p;
    long i = (long)blockIdx.x * blockDim.x + threadIdx.x;
    long stride = (long)gridDim.x * blockDim.x;
    const f32x4 z = {0.f, 0.f, 0.f, 0.f};
    for (; i < n4; i += stride) pg[i] = z;
}

// Scatter normalized histogram weights. Denominators are analytically constant:
// counts.sum == T exactly; recency sum is the geometric series (1-d^T)/(1-d).
__global__ void ucf_scatter_kernel(const int* __restrict__ X,
                                   const float* __restrict__ decay,
                                   float* __restrict__ freq_dist,
                                   float* __restrict__ rec_dist) {
    int i = blockIdx.x * blockDim.x + threadIdx.x;   // over B*T
    if (i >= B_ * T_) return;
    int b = i / T_;
    int t = i % T_;
    int id = ((const GLB int*)X)[i];
    float d = ((const GLB float*)decay)[0];
    float w  = __powf(d, (float)(T_ - 1 - t));
    float S  = (1.0f - __powf(d, (float)T_)) / (1.0f - d);
    atomicAdd(&freq_dist[(long)b * V_ + id], 1.0f / ((float)T_ + 1e-8f));
    atomicAdd(&rec_dist [(long)b * V_ + id], w / (S + 1e-8f));
}

// Tiled bf16 WMMA GEMM: Out[m,n] (+)= sum_k A[m,k] * W[n,k]
// Block = 256 threads = 8 waves; M-tile of 16 rows x full N=512 (4 N-tiles of
// 16 per wave). Grid.y = K-split; ATOMIC accumulates via atomic f32 add.
// Hot loop: 16 global_load_b128 batched in flight (q[16] live array, uniform
// base + u32 lane offsets), then perm-pack conversion + 4 WMMAs. A tile is
// double-buffered in LDS in WMMA fragment order; one barrier per step.
template <bool ATOMIC>
__global__ void ucf_gemm_bf16(const float* __restrict__ A_, int lda,
                              const float* __restrict__ W_, int ldw,
                              float* __restrict__ Out_, int ldo,
                              const float* __restrict__ bias_,
                              int K, int kChunk) {
    __shared__ __align__(32) unsigned short ldsA[2][16 * 32];

    const GLB float* A   = (const GLB float*)A_;
    const GLB float* W   = (const GLB float*)W_;
    GLB float*       Out = (GLB float*)Out_;

    const int tid  = threadIdx.x;
    const int lane = tid & 31;
    const int wave = tid >> 5;
    const int m0   = blockIdx.x * 16;
    const int k0   = blockIdx.y * kChunk;
    const int kEnd = (k0 + kChunk < K) ? (k0 + kChunk) : K;
    const int span  = kEnd - k0;
    const int nfull = span >> 5;   // branch-free 32-wide steps
    const int tail  = span & 31;   // leftover K (only last split of big GEMM)

    // --- fixed A-staging slot: this thread owns 2 consecutive K of one row ---
    // A fragment order used here: lane m holds K {0..15}; lane m+16 holds
    // K {16..31}, contiguous.
    const int ae   = tid * 2;
    const int am   = ae >> 5;          // row 0..15
    const int ak   = ae & 31;          // even k
    const int aseg = ak >> 3;
    const int aoff = (am + ((ak >> 4) ? 16 : 0)) * 16 +
                     (ak & 15);   // ushort index (even)
    // uniform base + u32 lane offset (A spans < 4G floats here)
    unsigned aofs = (unsigned)((m0 + am) * lda + k0 + ak);

    v8f acc[4];
#pragma unroll
    for (int t = 0; t < 4; ++t) acc[t] = (v8f){0.f,0.f,0.f,0.f,0.f,0.f,0.f,0.f};

    const int colLane = wave * 64 + (lane & 15);
    const int kHalf   = (lane >> 4) * 16;   // B frag: lanes 16-31 hold K 16..31
    unsigned wofs[4];                        // u32 float offsets into W
#pragma unroll
    for (int t = 0; t < 4; ++t)
        wofs[t] = (unsigned)((colLane + t * 16) * ldw + k0 + kHalf);

    if (nfull > 0) {   // stage step 0 (unguarded: full step)
        f32x2 v = *(const GLB f32x2*)(A + aofs);
        *reinterpret_cast<unsigned*>(&ldsA[0][aoff]) = pkbf(v.x, v.y);
    }

    // ---------------- branch-free hot loop ----------------
    for (int s = 0; s < nfull; ++s) {
        __syncthreads();
        const v16bf afrag =
            *reinterpret_cast<const v16bf*>(&ldsA[s & 1][lane * 16]);
        if (s + 1 < nfull) {   // single uniform compare; stage next buffer
            f32x2 v = *(const GLB f32x2*)(A + aofs + (s + 1) * 32);
            *reinterpret_cast<unsigned*>(&ldsA[(s + 1) & 1][aoff]) =
                pkbf(v.x, v.y);
        }

        // Batch ALL 16 B loads first: 16 global_load_b128 in flight per wave.
        f32x4 q[16];
#pragma unroll
        for (int t = 0; t < 4; ++t) {
            const GLB f32x4* p4 = (const GLB f32x4*)(W + wofs[t]);
            q[4*t + 0] = p4[0];
            q[4*t + 1] = p4[1];
            q[4*t + 2] = p4[2];
            q[4*t + 3] = p4[3];
            __builtin_prefetch((const float*)(W + wofs[t] + 32), 0, 3);
        }
        // Convert + WMMA per tile.
#pragma unroll
        for (int t = 0; t < 4; ++t) {
            union { unsigned u[8]; v16bf v; } bf;
            bf.u[0] = pkbf(q[4*t+0].x, q[4*t+0].y);
            bf.u[1] = pkbf(q[4*t+0].z, q[4*t+0].w);
            bf.u[2] = pkbf(q[4*t+1].x, q[4*t+1].y);
            bf.u[3] = pkbf(q[4*t+1].z, q[4*t+1].w);
            bf.u[4] = pkbf(q[4*t+2].x, q[4*t+2].y);
            bf.u[5] = pkbf(q[4*t+2].z, q[4*t+2].w);
            bf.u[6] = pkbf(q[4*t+3].x, q[4*t+3].y);
            bf.u[7] = pkbf(q[4*t+3].z, q[4*t+3].w);
            acc[t] = __builtin_amdgcn_wmma_f32_16x16x32_bf16(
                false, afrag, false, bf.v, (short)0, acc[t], false, false);
            wofs[t] += 32;
        }
    }

    // ---------------- peeled tail step (guarded) ----------------
    if (tail) {
        const int ks = k0 + nfull * 32;
        __syncthreads();   // all reads of ping-pong buffers are done
        {
            float vx = (ks + ak     < K) ? A[aofs + nfull * 32]     : 0.0f;
            float vy = (ks + ak + 1 < K) ? A[aofs + nfull * 32 + 1] : 0.0f;
            *reinterpret_cast<unsigned*>(&ldsA[nfull & 1][aoff]) = pkbf(vx, vy);
        }
        __syncthreads();
        const v16bf afrag =
            *reinterpret_cast<const v16bf*>(&ldsA[nfull & 1][lane * 16]);
#pragma unroll
        for (int t = 0; t < 4; ++t) {
            float w2[16];
#pragma unroll
            for (int j = 0; j < 16; ++j)
                w2[j] = (ks + kHalf + j < K) ? W[wofs[t] + j] : 0.0f;
            union { unsigned u[8]; v16bf v; } bf;
#pragma unroll
            for (int j = 0; j < 8; ++j) bf.u[j] = pkbf(w2[2*j], w2[2*j+1]);
            acc[t] = __builtin_amdgcn_wmma_f32_16x16x32_bf16(
                false, afrag, false, bf.v, (short)0, acc[t], false, false);
        }
    }

    // C layout: VGPR i, lanes 0-15 -> M=i, lanes 16-31 -> M=i+8; N = lane&15.
    const int rowOff = (lane >> 4) * 8;
#pragma unroll
    for (int t = 0; t < 4; ++t) {
        const int col = colLane + t * 16;
#pragma unroll
        for (int i = 0; i < 8; ++i) {
            const int row = m0 + i + rowOff;
            float v = acc[t][i];
            if (ATOMIC) {
                atomicAdd(&Out_[(long)row * ldo + col], v);
            } else {
                Out[(long)row * ldo + col] = v + ((const GLB float*)bias_)[col];
            }
        }
    }
}

// emb[m, z*D+d] = relu(acc_z[m,d] + bias_z[d]) with interleaved (B, 2D) output
__global__ void ucf_bias_relu_kernel(const float* __restrict__ acc,
                                     const float* __restrict__ fb,
                                     const float* __restrict__ rb,
                                     float* __restrict__ emb) {
    int i = blockIdx.x * blockDim.x + threadIdx.x; // over B*2D
    if (i >= B_ * 2 * D_) return;
    int m = i / (2 * D_);
    int c = i % (2 * D_);
    int z = c / D_;
    int d = c % D_;
    float v = ((const GLB float*)acc)[(long)z * B_ * D_ + (long)m * D_ + d] +
              (z ? ((const GLB float*)rb)[d] : ((const GLB float*)fb)[d]);
    ((GLB float*)emb)[i] = v > 0.0f ? v : 0.0f;
}

extern "C" void kernel_launch(void* const* d_in, const int* in_sizes, int n_in,
                              void* d_out, int out_size, void* d_ws, size_t ws_size,
                              hipStream_t stream) {
    const int*   X       = (const int*)d_in[0];
    const float* freq_W  = (const float*)d_in[1];
    const float* freq_b  = (const float*)d_in[2];
    const float* rec_W   = (const float*)d_in[3];
    const float* rec_b   = (const float*)d_in[4];
    const float* comb_W  = (const float*)d_in[5];
    const float* comb_b  = (const float*)d_in[6];
    const float* decay   = (const float*)d_in[7];

    float* out       = (float*)d_out;
    float* combined  = out;                                   // B*D
    float* freq_dist = out + (long)B_ * D_;                   // B*V
    float* rec_dist  = freq_dist + (long)B_ * V_;             // B*V

    float* wsf  = (float*)d_ws;
    float* accF = wsf;                        // B*D pre-bias accumulator (freq)
    float* accR = wsf + (long)B_ * D_;        // B*D pre-bias accumulator (recency)
    float* emb  = wsf + 2L * B_ * D_;         // B*2D relu'd, interleaved

    // 1) zero dist outputs (contiguous) and scratch accumulators
    ucf_zero_kernel<<<4096, 256, 0, stream>>>(freq_dist, (2L * B_ * V_) / 4);
    ucf_zero_kernel<<<128, 256, 0, stream>>>(accF, (2L * B_ * D_) / 4);

    // 2) normalized histogram scatter
    ucf_scatter_kernel<<<(B_ * T_ + 255) / 256, 256, 0, stream>>>(
        X, decay, freq_dist, rec_dist);

    // 3) big GEMMs: dist @ W.T with 16-way K-split, atomic f32 accumulation
    const int kChunk = 3136; // 16 * 3136 = 50176 >= V, multiple of 32
    dim3 gridBig(B_ / 16, 16);
    ucf_gemm_bf16<true><<<gridBig, 256, 0, stream>>>(
        freq_dist, V_, freq_W, V_, accF, D_, nullptr, V_, kChunk);
    ucf_gemm_bf16<true><<<gridBig, 256, 0, stream>>>(
        rec_dist, V_, rec_W, V_, accR, D_, nullptr, V_, kChunk);

    // 4) bias + relu, interleave into (B, 2D)
    ucf_bias_relu_kernel<<<(B_ * 2 * D_ + 255) / 256, 256, 0, stream>>>(
        accF, freq_b, rec_b, emb);

    // 5) combine GEMM: emb (128x1024) @ comb_W.T + comb_b, direct store
    dim3 gridC(B_ / 16, 1);
    ucf_gemm_bf16<false><<<gridC, 256, 0, stream>>>(
        emb, 2 * D_, comb_W, 2 * D_, combined, D_, comb_b, 2 * D_, 2 * D_);
}
